// AttentionDW_2714419331280
// MI455X (gfx1250) — matrix operand order repaired and checked by reading, not compiled, not run on hardware
//
#include <hip/hip_runtime.h>
#include <hip/hip_bf16.h>

typedef __attribute__((ext_vector_type(16))) __bf16 v16bf;
typedef __attribute__((ext_vector_type(8)))  __bf16 v8bf;
typedef __attribute__((ext_vector_type(8)))  float  v8f;

#define DIM    256
#define HEADS  4
#define HD     64
#define BATCH  8
#define HH     64
#define WW     64
#define TQ     (HH * WW)   /* 4096 */
#define TKV    (TQ / 4)    /* 1024 */
#define EPSV   1e-5f
#define ATTN_SCALE 0.0625f /* DIM^-0.5, folded into Q at projection time */

// ---------------------------------------------------------------------------
// WMMA fragment loaders following CDNA5 ISA VGPR layouts (CDNA5 ISA)
// A (16x32 bf16): lanes 0-15 row M=lane, K = {0..7, 16..23};
//                 lanes 16-31 row M=lane-16, K = {8..15, 24..31}
// B (32x16 bf16): lanes 0-15 col N=lane, K = {0..7, 16..23};
//                 lanes 16-31 col N=lane-16, K = {8..15, 24..31}
// Caller passes a pointer to this lane's row/col at K=0 (16-byte aligned).
// ---------------------------------------------------------------------------
__device__ __forceinline__ v16bf load_a_frag(const __bf16* row, int lane) {
  const int kb = (lane < 16) ? 0 : 8;
  v16bf a;
  v8bf* p = reinterpret_cast<v8bf*>(&a);
  p[0] = *reinterpret_cast<const v8bf*>(row + kb);
  p[1] = *reinterpret_cast<const v8bf*>(row + kb + 16);
  return a;
}

__device__ __forceinline__ v16bf load_b_frag(const __bf16* col, int lane) {
  const int kb = (lane < 16) ? 0 : 8;
  v16bf b;
  v8bf* p = reinterpret_cast<v8bf*>(&b);
  p[0] = *reinterpret_cast<const v8bf*>(col + kb);
  p[1] = *reinterpret_cast<const v8bf*>(col + kb + 16);
  return b;
}

__device__ __forceinline__ v8f wmma_bf16(v16bf a, v16bf b, v8f c) {
  return __builtin_amdgcn_wmma_f32_16x16x32_bf16(false, a, false, b,
                                                 (short)0, c, false, false);
}

// ---------------------------------------------------------------------------
// Stage 1: depthwise 3x3 conv + BatchNorm (inference) -> bf16 y[b][c][t]
// Tiny FLOP share; x re-reads hit L2 (33 MB << 192 MB).
// ---------------------------------------------------------------------------
__global__ __launch_bounds__(256, 4)
void dwbn_kernel(const float* __restrict__ x,
                 const float* __restrict__ dw,
                 const float* __restrict__ gam,
                 const float* __restrict__ bta,
                 const float* __restrict__ mu,
                 const float* __restrict__ var,
                 __bf16* __restrict__ y,
                 int stride, int Ho, int Wo) {
  int idx = blockIdx.x * blockDim.x + threadIdx.x;
  int total = BATCH * DIM * Ho * Wo;
  if (idx >= total) return;
  int ox = idx % Wo;
  int oy = (idx / Wo) % Ho;
  int c  = (idx / (Wo * Ho)) % DIM;
  int b  = idx / (Wo * Ho * DIM);
  const float* xp = x + (size_t)(b * DIM + c) * (HH * WW);
  const float* wp = dw + c * 9;
  float acc = 0.f;
  int iy0 = oy * stride - 1, ix0 = ox * stride - 1;
#pragma unroll
  for (int ky = 0; ky < 3; ++ky) {
    int iy = iy0 + ky;
    if ((unsigned)iy < (unsigned)HH) {
#pragma unroll
      for (int kx = 0; kx < 3; ++kx) {
        int ix = ix0 + kx;
        if ((unsigned)ix < (unsigned)WW)
          acc += xp[iy * WW + ix] * wp[ky * 3 + kx];
      }
    }
  }
  float inv = gam[c] * __frsqrt_rn(var[c] + EPSV);
  y[idx] = (__bf16)((acc - mu[c]) * inv + bta[c]);
}

// ---------------------------------------------------------------------------
// Stage 2/4: 256xT x 256 GEMM, bf16 WMMA, f32 accumulate.
// Workgroup tile 128(M) x 64(N), 8 waves (4x2), 2x2 16x16 accs per wave,
// K staged through LDS in 32-wide slabs (padded stride 40 halfwords).
// MODE 0: bf16 out [c][t]            (V)
// MODE 1: bf16 out [t][c], * scale   (Q with attn scale folded in, K)
// MODE 2: f32 out [c][t] + bias      (final projection into d_out)
// ---------------------------------------------------------------------------
template <int MODE>
__global__ __launch_bounds__(256, 1)
void gemm_wmma_kernel(const float* __restrict__ W,   // [DIM][DIM] f32
                      const __bf16* __restrict__ Y,  // [B][DIM][T] bf16
                      void* __restrict__ Out,
                      const float* __restrict__ bias,
                      int T, float outScale) {
  __shared__ __bf16 Als[128 * 40];
  __shared__ __bf16 Bls[64 * 40];
  const int tid   = threadIdx.x;
  const int lane  = tid & 31;
  const int wave  = tid >> 5;   // 0..7
  const int wm    = wave & 3;   // 32 M-rows per wave
  const int wn    = wave >> 2;  // 32 N-cols per wave
  const int nBase = blockIdx.x * 64;
  const int mBase = blockIdx.y * 128;
  const int b     = blockIdx.z;
  const __bf16* Yb = Y + (size_t)b * DIM * T;

  v8f acc[2][2] = {};

  for (int k0 = 0; k0 < DIM; k0 += 32) {
    // Stage A: weights (f32 -> bf16 on the fly)
    {
      int kk = tid & 31, mm = tid >> 5;
#pragma unroll
      for (int j = 0; j < 16; ++j)
        Als[(mm + 8 * j) * 40 + kk] =
            (__bf16)W[(size_t)(mBase + mm + 8 * j) * DIM + k0 + kk];
    }
    // Stage B transposed: Bls[n][k] = Y[k0+k][nBase+n]
    {
      int nn = tid & 63, kc = tid >> 6;
#pragma unroll
      for (int j = 0; j < 8; ++j) {
        int k = kc * 8 + j;
        Bls[nn * 40 + k] = Yb[(size_t)(k0 + k) * T + nBase + nn];
      }
    }
    __syncthreads();

    v16bf af[2], bf[2];
#pragma unroll
    for (int i = 0; i < 2; ++i)
      af[i] = load_a_frag(&Als[(wm * 32 + i * 16 + (lane & 15)) * 40], lane);
#pragma unroll
    for (int j = 0; j < 2; ++j)
      bf[j] = load_b_frag(&Bls[(wn * 32 + j * 16 + (lane & 15)) * 40], lane);
#pragma unroll
    for (int i = 0; i < 2; ++i)
#pragma unroll
      for (int j = 0; j < 2; ++j)
        acc[i][j] = wmma_bf16(af[i], bf[j], acc[i][j]);
    __syncthreads();
  }

  // C layout: VGPR r -> M = r (lanes 0-15) / r+8 (lanes 16-31), N = lane&15
#pragma unroll
  for (int i = 0; i < 2; ++i)
#pragma unroll
    for (int j = 0; j < 2; ++j) {
      int Mb = mBase + wm * 32 + i * 16 + ((lane < 16) ? 0 : 8);
      int N  = nBase + wn * 32 + j * 16 + (lane & 15);
#pragma unroll
      for (int r = 0; r < 8; ++r) {
        int M = Mb + r;
        float v = acc[i][j][r];
        if (MODE == 0)
          ((__bf16*)Out)[(size_t)b * DIM * T + (size_t)M * T + N] = (__bf16)v;
        else if (MODE == 1)
          ((__bf16*)Out)[(size_t)b * DIM * T + (size_t)N * DIM + M] =
              (__bf16)(v * outScale);
        else
          ((float*)Out)[(size_t)b * DIM * T + (size_t)M * T + N] = v + bias[M];
      }
    }
}

// ---------------------------------------------------------------------------
// Stage 3: flash attention per (b, head). Block = 4 waves x 16 query rows.
// KV swept in 64-wide chunks: 8 WMMA for S = (Q*scale) K^T, online softmax
// (row stats lane-local after a 16-lane shfl_xor tree, courtesy of the C-tile
// layout), P re-laid-out through a per-wave LDS tile, 8 WMMA for O += P V.
// The DIM^-0.5 scale is pre-folded into Q, so no per-chunk scaling here.
// ---------------------------------------------------------------------------
__global__ __launch_bounds__(128, 1)
void attn_wmma_kernel(const __bf16* __restrict__ Qt,  // [B][TQ][DIM] (pre-scaled)
                      const __bf16* __restrict__ Kt,  // [B][TKV][DIM]
                      const __bf16* __restrict__ Vc,  // [B][DIM][TKV]
                      __bf16* __restrict__ Oc) {      // [B][DIM][TQ]
  __shared__ __bf16 Pls[4 * 16 * 72];
  const int lane  = threadIdx.x & 31;
  const int wave  = threadIdx.x >> 5;             // 0..3
  const int lBase = blockIdx.x * 64 + wave * 16;  // query-row base for wave
  const int head  = blockIdx.y;
  const int b     = blockIdx.z;
  const int cBase = head * HD;

  // Persistent Q fragments: rows lBase..lBase+15, K = d (0..63)
  const __bf16* qrow =
      Qt + ((size_t)b * TQ + lBase + (lane & 15)) * DIM + cBase;
  const v16bf qf0 = load_a_frag(qrow, lane);
  const v16bf qf1 = load_a_frag(qrow + 32, lane);

  float Mrow[8], Lrow[8];
#pragma unroll
  for (int r = 0; r < 8; ++r) { Mrow[r] = -1e30f; Lrow[r] = 0.f; }
  v8f O[4] = {};

  __bf16* P = &Pls[wave * 16 * 72];  // per-wave 16x64 tile, stride 72

  const __bf16* kchunk = Kt + ((size_t)b * TKV + (lane & 15)) * DIM + cBase;
  const __bf16* vchunk = Vc + ((size_t)b * DIM + cBase + (lane & 15)) * TKV;

  for (int kv = 0; kv < TKV; kv += 64) {
    // ---- S = Q K^T over this 64-wide KV chunk (4 N-subtiles) ----
    v8f S[4] = {};
#pragma unroll
    for (int sub = 0; sub < 4; ++sub) {
      const __bf16* krow = kchunk + (size_t)(kv + sub * 16) * DIM;
      v16bf kf0 = load_b_frag(krow, lane);
      v16bf kf1 = load_b_frag(krow + 32, lane);
      S[sub] = wmma_bf16(qf0, kf0, S[sub]);
      S[sub] = wmma_bf16(qf1, kf1, S[sub]);
    }

    // ---- online softmax (row-wise across 64 cols: 4 tiles + 16 lanes) ----
#pragma unroll
    for (int r = 0; r < 8; ++r) {
      float mx = fmaxf(fmaxf(S[0][r], S[1][r]), fmaxf(S[2][r], S[3][r]));
#pragma unroll
      for (int m = 8; m >= 1; m >>= 1) mx = fmaxf(mx, __shfl_xor(mx, m, 32));
      float Mnew  = fmaxf(Mrow[r], mx);
      float alpha = __expf(Mrow[r] - Mnew);
      Mrow[r] = Mnew;
      float rs = 0.f;
#pragma unroll
      for (int sub = 0; sub < 4; ++sub) {
        float e = __expf(S[sub][r] - Mnew);
        S[sub][r] = e;
        rs += e;
      }
#pragma unroll
      for (int m = 8; m >= 1; m >>= 1) rs += __shfl_xor(rs, m, 32);
      Lrow[r] = Lrow[r] * alpha + rs;
#pragma unroll
      for (int sub = 0; sub < 4; ++sub) O[sub][r] *= alpha;
    }

    // ---- re-layout P: C-tile layout -> LDS -> A fragments ----
    {
      const int hi8 = (lane < 16) ? 0 : 8;
#pragma unroll
      for (int sub = 0; sub < 4; ++sub)
#pragma unroll
        for (int r = 0; r < 8; ++r)
          P[(r + hi8) * 72 + sub * 16 + (lane & 15)] = (__bf16)S[sub][r];
    }
    v16bf pf0 = load_a_frag(&P[(lane & 15) * 72], lane);
    v16bf pf1 = load_a_frag(&P[(lane & 15) * 72 + 32], lane);

    // ---- O += P V ----
#pragma unroll
    for (int sub = 0; sub < 4; ++sub) {
      const __bf16* vrow = vchunk + (size_t)sub * 16 * TKV + kv;
      v16bf vf0 = load_b_frag(vrow, lane);
      v16bf vf1 = load_b_frag(vrow + 32, lane);
      O[sub] = wmma_bf16(pf0, vf0, O[sub]);
      O[sub] = wmma_bf16(pf1, vf1, O[sub]);
    }
  }

  // ---- normalize and emit as bf16 [b][head*64+d][t] for the proj GEMM ----
#pragma unroll
  for (int r = 0; r < 8; ++r) {
    float inv = 1.f / Lrow[r];
    int t = lBase + r + ((lane < 16) ? 0 : 8);
#pragma unroll
    for (int sub = 0; sub < 4; ++sub) {
      int c = cBase + sub * 16 + (lane & 15);
      Oc[((size_t)b * DIM + c) * TQ + t] = (__bf16)(O[sub][r] * inv);
    }
  }
}

// ---------------------------------------------------------------------------
extern "C" void kernel_launch(void* const* d_in, const int* in_sizes, int n_in,
                              void* d_out, int out_size, void* d_ws, size_t ws_size,
                              hipStream_t stream) {
  const float* x = (const float*)d_in[0];
  // per-projection params: dw, g, b, m, v, pw
  const float* dwq = (const float*)d_in[1];
  const float* gq  = (const float*)d_in[2];
  const float* bq  = (const float*)d_in[3];
  const float* mq  = (const float*)d_in[4];
  const float* vq  = (const float*)d_in[5];
  const float* pwq = (const float*)d_in[6];
  const float* dwk = (const float*)d_in[7];
  const float* gk  = (const float*)d_in[8];
  const float* bk  = (const float*)d_in[9];
  const float* mk  = (const float*)d_in[10];
  const float* vk  = (const float*)d_in[11];
  const float* pwk = (const float*)d_in[12];
  const float* dwv = (const float*)d_in[13];
  const float* gv  = (const float*)d_in[14];
  const float* bv  = (const float*)d_in[15];
  const float* mv  = (const float*)d_in[16];
  const float* vv  = (const float*)d_in[17];
  const float* pwv = (const float*)d_in[18];
  const float* pjw = (const float*)d_in[19];
  const float* pjb = (const float*)d_in[20];
  float* out = (float*)d_out;

  // workspace carve-up (bf16), ~64 MB total
  const size_t szQ = (size_t)BATCH * DIM * TQ;   // 8M elems
  const size_t szK = (size_t)BATCH * DIM * TKV;  // 2M elems
  __bf16* p  = (__bf16*)d_ws;
  __bf16* yq = p; p += szQ;
  __bf16* yk = p; p += szK;
  __bf16* yv = p; p += szK;
  __bf16* qt = p; p += szQ;   // Q transposed  [b][t][c], pre-scaled
  __bf16* kt = p; p += szK;   // K transposed  [b][t][c]
  __bf16* vb = p; p += szK;   // V             [b][c][t]
  __bf16* ao = p; p += szQ;   // attention out [b][c][t]

  // Stage 1: depthwise + BN
  int totQ = BATCH * DIM * TQ;
  int totK = BATCH * DIM * TKV;
  dwbn_kernel<<<(totQ + 255) / 256, 256, 0, stream>>>(x, dwq, gq, bq, mq, vq, yq, 1, 64, 64);
  dwbn_kernel<<<(totK + 255) / 256, 256, 0, stream>>>(x, dwk, gk, bk, mk, vk, yk, 2, 32, 32);
  dwbn_kernel<<<(totK + 255) / 256, 256, 0, stream>>>(x, dwv, gv, bv, mv, vv, yv, 2, 32, 32);

  // Stage 2: pointwise projections (bf16 WMMA GEMMs); attn scale folded into Q
  gemm_wmma_kernel<1><<<dim3(TQ / 64, 2, BATCH), 256, 0, stream>>>(
      pwq, yq, qt, nullptr, TQ, ATTN_SCALE);
  gemm_wmma_kernel<1><<<dim3(TKV / 64, 2, BATCH), 256, 0, stream>>>(
      pwk, yk, kt, nullptr, TKV, 1.0f);
  gemm_wmma_kernel<0><<<dim3(TKV / 64, 2, BATCH), 256, 0, stream>>>(
      pwv, yv, vb, nullptr, TKV, 1.0f);

  // Stage 3: flash attention
  attn_wmma_kernel<<<dim3(TQ / 64, HEADS, BATCH), 128, 0, stream>>>(qt, kt, vb, ao);

  // Stage 4: output projection + bias, f32 into d_out
  gemm_wmma_kernel<2><<<dim3(TQ / 64, 2, BATCH), 256, 0, stream>>>(
      pjw, ao, out, pjb, TQ, 1.0f);
}
